// TemporalNeighborhoodAttention_87557203297015
// MI455X (gfx1250) — matrix operand order repaired and checked by reading, not compiled, not run on hardware
//
#include <hip/hip_runtime.h>
#include <hip/hip_bf16.h>

typedef __attribute__((ext_vector_type(16))) __bf16 v16bf;
typedef __attribute__((ext_vector_type(8)))  float  v8f;

#define B_DIM 4
#define T_DIM 4096
#define C_DIM 512
#define H_DIM 8
#define D_DIM 64
#define KSZ 9
#define PADK 4

// ---------- helpers ----------

__device__ __forceinline__ unsigned int pack2_bf16(float x, float y) {
  // round-to-nearest-even f32 -> bf16, packed pair (low = x, high = y)
  unsigned int ux = __float_as_uint(x);
  unsigned int uy = __float_as_uint(y);
  ux = (ux + 0x7FFFu + ((ux >> 16) & 1u)) >> 16;
  uy = (uy + 0x7FFFu + ((uy >> 16) & 1u)) >> 16;
  return (ux & 0xFFFFu) | (uy << 16);
}

union FragBF { unsigned int u[8]; v16bf v; };

// CDNA5 async DMA: global -> LDS, 16 bytes per lane, tracked by ASYNCcnt.
// VDST operand carries the LDS byte address (low 32 bits of generic ptr).
__device__ __forceinline__ void async_load_b128(unsigned int ldsByteOff,
                                                const void* gptr) {
  asm volatile("global_load_async_to_lds_b128 %0, %1, off"
               :: "v"(ldsByteOff), "v"((unsigned long long)(uintptr_t)gptr)
               : "memory");
}

__device__ __forceinline__ void wait_asynccnt0() {
  asm volatile("s_wait_asynccnt 0x0" ::: "memory");
}

// ---------- f32 -> packed bf16 conversion pass ----------

__global__ __launch_bounds__(256)
void f32_to_bf16_kernel(const float* __restrict__ src,
                        unsigned int* __restrict__ dst, int npairs) {
  const int i = blockIdx.x * blockDim.x + threadIdx.x;
  if (i < npairs) {
    const float2 v = ((const float2*)src)[i];
    dst[i] = pack2_bf16(v.x, v.y);
  }
}

// ---------- bf16 WMMA GEMM with async-LDS double buffering ----------
// Out[m,n] = sum_k A[m,k] * W[n,k] + bias[n]
// A: M x 512 bf16 row-major, W: 512(N) x 512(K) bf16 row-major.
// Block tile 128(M) x 64(N), BK=32, 256 threads = 8 waves, wave tile 32x32.
// LDS rows padded to 80 bytes (20 dwords): 16B-aligned DMA, conflict-free ds.

#define LDS_ROW_U 20            // dwords per 32-element bf16 row (16 + 4 pad)
#define LDS_ROW_B 80            // bytes
#define A_ROWS 128
#define B_ROWS 64
#define BUF_U (A_ROWS * LDS_ROW_U + B_ROWS * LDS_ROW_U)   // 3840 dwords
#define B_OFF_U (A_ROWS * LDS_ROW_U)                      // 2560 dwords
#define BUF_BYTES (BUF_U * 4)                             // 15360 bytes

template <bool HEAD_SPLIT>
__global__ __launch_bounds__(256)
void gemm_wmma_async(const unsigned short* __restrict__ Abf,
                     const unsigned short* __restrict__ W0,
                     const unsigned short* __restrict__ W1,
                     const unsigned short* __restrict__ W2,
                     const float* __restrict__ b0, const float* __restrict__ b1,
                     const float* __restrict__ b2,
                     float* __restrict__ O0, float* __restrict__ O1,
                     float* __restrict__ O2) {
  __shared__ __align__(16) unsigned int smem[2 * BUF_U];

  const int z = blockIdx.z;
  const unsigned short* W = (z == 0) ? W0 : ((z == 1) ? W1 : W2);
  const float* bias       = (z == 0) ? b0 : ((z == 1) ? b1 : b2);
  float* Out              = (z == 0) ? O0 : ((z == 1) ? O1 : O2);

  const int tid  = threadIdx.x;
  const int lane = tid & 31;
  const int wid  = tid >> 5;
  const int mBase = blockIdx.x * 128;
  const int nBase = blockIdx.y * 64;
  const int rm0 = (wid & 3) * 32;   // wave M offset within tile (2 subtiles)
  const int cn  = (wid >> 2) * 32;  // wave N offset within tile (2 subtiles)

  // LDS byte base of smem = low 32 bits of its generic address (LDS aperture).
  const unsigned int smemByte = (unsigned int)(uintptr_t)&smem[0];

  // DMA assignments: A tile 128x32 bf16 = 512 b128 ops (2/thread),
  //                  B tile  64x32 bf16 = 256 b128 ops (1/thread).
  const int arow0 = tid >> 2;            // 0..63
  const int arow1 = arow0 + 64;          // 64..127
  const int aseg  = tid & 3;             // 16B segment within row
  const unsigned short* aG0 = Abf + (size_t)(mBase + arow0) * C_DIM + aseg * 8;
  const unsigned short* aG1 = Abf + (size_t)(mBase + arow1) * C_DIM + aseg * 8;
  const unsigned short* bG  = W   + (size_t)(nBase + arow0) * C_DIM + aseg * 8;
  const unsigned int aL0 = (unsigned int)(arow0 * LDS_ROW_B + aseg * 16);
  const unsigned int aL1 = (unsigned int)(arow1 * LDS_ROW_B + aseg * 16);
  const unsigned int bL  = (unsigned int)(B_OFF_U * 4 + arow0 * LDS_ROW_B + aseg * 16);

  auto issue_tile = [&](int buf, int kb) {
    const unsigned int base = smemByte + (unsigned int)(buf * BUF_BYTES);
    async_load_b128(base + aL0, aG0 + kb);
    async_load_b128(base + aL1, aG1 + kb);
    async_load_b128(base + bL,  bG  + kb);
  };

  v8f c00 = {}, c01 = {}, c10 = {}, c11 = {};

  issue_tile(0, 0);
  for (int i = 0; i < C_DIM / 32; ++i) {
    wait_asynccnt0();     // this wave's DMA for buffer i&1 landed in LDS
    __syncthreads();      // everyone's DMA landed
    if (i + 1 < C_DIM / 32) issue_tile((i + 1) & 1, (i + 1) * 32);

    const unsigned int* As = &smem[(i & 1) * BUF_U];
    const unsigned int* Bs = As + B_OFF_U;

    // A fragments (16x32 bf16): lanes 0-15 K={0..7,16..23}, 16-31 K={8..15,24..31}
    FragBF fa0, fa1;
    const int ar = rm0 + (lane & 15);
    const int ksel = (lane < 16) ? 0 : 4;
#pragma unroll
    for (int j = 0; j < 8; ++j) {
      const int kk = ((j < 4) ? j : (j + 4)) + ksel;
      fa0.u[j] = As[ar * LDS_ROW_U + kk];
      fa1.u[j] = As[(ar + 16) * LDS_ROW_U + kk];
    }
    // B fragments (32x16 bf16): lane holds column n=lane%16; lanes 0-15 K={0..7,16..23}, 16-31 K={8..15,24..31}
    FragBF fb0, fb1;
    const int hi = (lane < 16) ? 0 : 4;
    const int br = cn + (lane & 15);
#pragma unroll
    for (int j = 0; j < 8; ++j) {
      const int kb2 = ((j < 4) ? j : (j + 4)) + hi;
      fb0.u[j] = Bs[br * LDS_ROW_U + kb2];
      fb1.u[j] = Bs[(br + 16) * LDS_ROW_U + kb2];
    }
    c00 = __builtin_amdgcn_wmma_f32_16x16x32_bf16(false, fa0.v, false, fb0.v,
                                                  (short)0, c00, false, false);
    c01 = __builtin_amdgcn_wmma_f32_16x16x32_bf16(false, fa0.v, false, fb1.v,
                                                  (short)0, c01, false, false);
    c10 = __builtin_amdgcn_wmma_f32_16x16x32_bf16(false, fa1.v, false, fb0.v,
                                                  (short)0, c10, false, false);
    c11 = __builtin_amdgcn_wmma_f32_16x16x32_bf16(false, fa1.v, false, fb1.v,
                                                  (short)0, c11, false, false);
    __syncthreads();      // reads done before this buffer is re-DMA'd
  }

  // C layout: VGPR r -> M = r (lanes 0-15) / r+8 (lanes 16-31), N = lane%16.
  const int ncol0 = nBase + cn + (lane & 15);
#pragma unroll
  for (int mi = 0; mi < 2; ++mi) {
    const int mrow0 = mBase + rm0 + mi * 16 + ((lane >> 4) << 3);
#pragma unroll
    for (int nt = 0; nt < 2; ++nt) {
      const int ncol = ncol0 + nt * 16;
      const float bval = bias[ncol];
      v8f cc = mi ? (nt ? c11 : c10) : (nt ? c01 : c00);
#pragma unroll
      for (int r = 0; r < 8; ++r) {
        const int m = mrow0 + r;
        const float val = cc[r] + bval;
        if (HEAD_SPLIT) {
          const int bb = m >> 12;           // m / T
          const int tt = m & (T_DIM - 1);
          const int hh = ncol >> 6;         // n / D
          const int dd = ncol & 63;
          Out[(((size_t)bb * H_DIM + hh) * T_DIM + tt) * D_DIM + dd] = val;
        } else {
          Out[(size_t)m * C_DIM + ncol] = val;
        }
      }
    }
  }
}

// ---------- neighborhood attention: one wave32 per (b,h,t) ----------
// q,k,v: f32 (B,H,T,D). Output written as packed bf16 (B,T,C) for the
// output-projection GEMM.

__global__ __launch_bounds__(128)
void neigh_attn_kernel(const float* __restrict__ q, const float* __restrict__ k,
                       const float* __restrict__ v,
                       unsigned int* __restrict__ ybf /* (B,T,C)/2 pairs */) {
  const int lane = threadIdx.x & 31;
  const int w = blockIdx.x * 4 + (threadIdx.x >> 5);  // flat (b*H+h)*T + t
  const int t  = w & (T_DIM - 1);
  const int bh = w >> 12;
  const int h  = bh & (H_DIM - 1);
  const int b  = bh >> 3;

  const size_t rowBase = (size_t)bh * T_DIM;
  const float2 qv = ((const float2*)(q + (rowBase + t) * D_DIM))[lane];

  const float scale = 0.125f;  // 1/sqrt(64)
  float e[KSZ];
  float mx = -3.4e38f;
#pragma unroll
  for (int kk = 0; kk < KSZ; ++kk) {
    int tk = t - PADK + kk;
    tk = tk < 0 ? 0 : (tk >= T_DIM ? T_DIM - 1 : tk);
    const float2 kv = ((const float2*)(k + (rowBase + tk) * D_DIM))[lane];
    float p = qv.x * kv.x + qv.y * kv.y;
#pragma unroll
    for (int off = 16; off > 0; off >>= 1) p += __shfl_xor(p, off, 32);
    p *= scale;
    e[kk] = p;
    mx = fmaxf(mx, p);
  }
  float s = 0.0f;
#pragma unroll
  for (int kk = 0; kk < KSZ; ++kk) {
    e[kk] = __expf(e[kk] - mx);
    s += e[kk];
  }
  const float inv = 1.0f / s;

  float2 acc = {0.0f, 0.0f};
#pragma unroll
  for (int kk = 0; kk < KSZ; ++kk) {
    int tk = t - PADK + kk;
    tk = tk < 0 ? 0 : (tk >= T_DIM ? T_DIM - 1 : tk);
    const float2 vv = ((const float2*)(v + (rowBase + tk) * D_DIM))[lane];
    const float a = e[kk] * inv;
    acc.x += a * vv.x;
    acc.y += a * vv.y;
  }
  // bf16 pair store: element offset (b*T+t)*C + h*64 + 2*lane
  const size_t pairIdx = (((size_t)(b * T_DIM + t) * C_DIM) >> 1) +
                         (unsigned)(h * (D_DIM / 2) + lane);
  ybf[pairIdx] = pack2_bf16(acc.x, acc.y);
}

// ---------- host-side orchestration ----------

extern "C" void kernel_launch(void* const* d_in, const int* in_sizes, int n_in,
                              void* d_out, int out_size, void* d_ws, size_t ws_size,
                              hipStream_t stream) {
  (void)in_sizes; (void)n_in; (void)out_size; (void)ws_size;
  const float* x  = (const float*)d_in[0];
  const float* Wq = (const float*)d_in[1];
  const float* bq = (const float*)d_in[2];
  const float* Wk = (const float*)d_in[3];
  const float* bk = (const float*)d_in[4];
  const float* Wv = (const float*)d_in[5];
  const float* bv = (const float*)d_in[6];
  const float* Wo = (const float*)d_in[7];
  const float* bo = (const float*)d_in[8];

  const size_t elems = (size_t)B_DIM * T_DIM * C_DIM;   // 8,388,608
  const size_t welem = (size_t)C_DIM * C_DIM;           // 262,144

  float* qws = (float*)d_ws;                    // (B,H,T,D) f32, 32 MB
  float* kws = qws + elems;                     // 32 MB
  float* vws = kws + elems;                     // 32 MB
  unsigned short* ybf = (unsigned short*)(vws + elems);     // (B,T,C) bf16, 16 MB
  unsigned short* xbf = ybf + elems;                        // (B*T,C) bf16, 16 MB
  unsigned short* wqb = xbf + elems;                        // 4 x 0.5 MB
  unsigned short* wkb = wqb + welem;
  unsigned short* wvb = wkb + welem;
  unsigned short* wob = wvb + welem;

  // One-pass f32 -> bf16 conversion of activations and weights.
  f32_to_bf16_kernel<<<(int)(elems / 2 / 256), 256, 0, stream>>>(
      x, (unsigned int*)xbf, (int)(elems / 2));
  f32_to_bf16_kernel<<<(int)(welem / 2 / 256), 256, 0, stream>>>(
      Wq, (unsigned int*)wqb, (int)(welem / 2));
  f32_to_bf16_kernel<<<(int)(welem / 2 / 256), 256, 0, stream>>>(
      Wk, (unsigned int*)wkb, (int)(welem / 2));
  f32_to_bf16_kernel<<<(int)(welem / 2 / 256), 256, 0, stream>>>(
      Wv, (unsigned int*)wvb, (int)(welem / 2));
  f32_to_bf16_kernel<<<(int)(welem / 2 / 256), 256, 0, stream>>>(
      Wo, (unsigned int*)wob, (int)(welem / 2));

  // Fused QKV projection: grid.z selects {Wq,Wk,Wv}; scatter to (B,H,T,D).
  dim3 gq(128, 8, 3);  // M/128 = 16384/128, N/64 = 512/64
  gemm_wmma_async<true><<<gq, 256, 0, stream>>>(xbf, wqb, wkb, wvb, bq, bk, bv,
                                                qws, kws, vws);

  // Neighborhood attention (f32), output packed bf16 in (B,T,C).
  neigh_attn_kernel<<<(B_DIM * H_DIM * T_DIM) / 4, 128, 0, stream>>>(
      qws, kws, vws, (unsigned int*)ybf);

  // Output projection into d_out (f32).
  dim3 go(128, 8, 1);
  gemm_wmma_async<false><<<go, 256, 0, stream>>>(ybf, wob, wob, wob, bo, bo, bo,
                                                 (float*)d_out, (float*)d_out,
                                                 (float*)d_out);
}
